// TopkRouter_17136919511683
// MI455X (gfx1250) — compile-verified, matrix operand order checked by reading, not run on hardware
//
#include <hip/hip_runtime.h>
#include <hip/hip_bf16.h>

// ---------------------------------------------------------------------------
// Top-2 MoE router, fused:  x[32768,1024] -> relu(x@W1+b1)[.,2048] -> @W2+b2
// -> logits[.,64] -> (softmax(l/0.01), sparse-top2-softmax, indices)
//
// bf16 WMMA (v_wmma_f32_16x16x32_bf16) with f32 accumulate. Weights are
// pre-packed into WMMA B-fragment layout in workspace (read-once, L2-resident:
// 4.25 MB << 192 MB L2). 64 tokens/WG (4 independent waves x 16 tokens) so W1
// is swept only 512 times (~2 GB L2 traffic) instead of 2048. x and outputs
// are non-temporal (streamed once) to keep L2 for the weights.
// ---------------------------------------------------------------------------

typedef __attribute__((ext_vector_type(16))) __bf16 v16bf;
typedef __attribute__((ext_vector_type(4)))  __bf16 v4bf;
typedef __attribute__((ext_vector_type(8)))  float  v8f;
typedef __attribute__((ext_vector_type(4)))  float  v4f;

#define DD    1024      // D
#define FF    2048      // 2D
#define EE    64        // experts
#define NTOK  32768     // B*S
#define INV_TEMP 100.0f

// Map K-within-chunk (0..31) to (lane-half, element index) per the CDNA5
// 16-bit A-matrix 16x32 VGPR layout (ISA 7.12.2).
__device__ __forceinline__ void kin2pos(int kin, int& half, int& idx) {
  if (kin < 8)       { half = 0; idx = kin; }
  else if (kin < 16) { half = 1; idx = kin - 8; }
  else if (kin < 24) { half = 0; idx = 8 + (kin - 16); }
  else               { half = 1; idx = 8 + (kin - 24); }
}

// Pack a row-major f32 weight [K, N] into bf16 WMMA B-fragments:
// fragment (kc, ft) holds the 32x16 tile K=kc*32.., N=ft*16.. ; per lane 16
// contiguous bf16 (32 bytes) so the GEMM loads it with two global_load_b128.
__global__ void pack_w_bf16(const float* __restrict__ w, __bf16* __restrict__ dst,
                            int K, int N, int ftiles) {
  int idx = blockIdx.x * blockDim.x + threadIdx.x;
  if (idx >= K * N) return;
  int k = idx / N, f = idx % N;
  int kc = k >> 5, kin = k & 31;
  int ft = f >> 4, n = f & 15;
  int half, i; kin2pos(kin, half, i);
  int lane = half * 16 + n;
  dst[(((size_t)kc * ftiles + ft) * 32 + lane) * 16 + i] = (__bf16)w[idx];
}

__global__ __launch_bounds__(128)
void router_fused(const float* __restrict__ x,
                  const float* __restrict__ bias1,
                  const float* __restrict__ bias2,
                  const __bf16* __restrict__ w1f,   // [32 kc][128 ft][32 lane][16]
                  const __bf16* __restrict__ w2f,   // [64 kc][4 ft][32 lane][16]
                  float* __restrict__ outOri,
                  float* __restrict__ outSp,
                  int*   __restrict__ outIdx) {
  // Per-wave x slab in A-fragment layout: 32 kc x 32 lanes x 16 bf16 = 32 KB.
  // Reused as float logits scratch (4 KB needed) in the epilogue.
  __shared__ __align__(32) __bf16 xA[4][32 * 32 * 16];
  // Per-wave h re-swizzle buffer: 32 lanes x 16 bf16 = 1 KB each
  __shared__ __align__(32) __bf16 hA[4][32 * 16];

  const int tid  = threadIdx.x;
  const int wave = tid >> 5;
  const int lane = tid & 31;
  const int tokw = blockIdx.x * 64 + wave * 16;   // this wave's 16 tokens

  __bf16* myxA = &xA[4 > 1 ? wave : 0][0];

  // ---- Stage x[16,1024] -> bf16 A-fragment swizzle in this wave's LDS slab.
  // b128 global loads (non-temporal: x is streamed once), packed b64 LDS
  // stores: a 4-aligned k quad never straddles a fragment half-boundary.
  for (int j = 0; j < 128; ++j) {
    int e  = (lane + 32 * j) * 4;       // flat element, 4-wide
    int m  = e >> 10;                   // token row 0..15
    int k0 = e & (DD - 1);              // feature 0..1020, multiple of 4
    v4f v = __builtin_nontemporal_load(
        (const v4f*)&x[(size_t)(tokw + m) * DD + k0]);
    int kc = k0 >> 5, kin = k0 & 31;
    int half, i; kin2pos(kin, half, i);
    v4bf b;
    b.x = (__bf16)v.x; b.y = (__bf16)v.y; b.z = (__bf16)v.z; b.w = (__bf16)v.w;
    *(v4bf*)&myxA[((kc * 32) + (half * 16 + m)) * 16 + i] = b;
  }
  // No barrier needed: each wave only touches its own slab; DS ops are
  // in-order per wave and the compiler inserts s_wait_dscnt for the RAW.

  // ---- Persistent logits accumulators: 4 N-tiles of 16 -> [16 x 64]
  v8f acc0 = {}, acc1 = {}, acc2 = {}, acc3 = {};

  const int n     = lane & 15;
  const int mbase = 8 * (lane >> 4);
  int h0, i0, h1, i1;
  kin2pos(n, h0, i0);        // where h column n    lands in the A-fragment
  kin2pos(16 + n, h1, i1);   // where h column 16+n lands in the A-fragment

  // Every wave sweeps the full 2048-wide hidden dim (64 blocks of 32); the
  // 4 waves walk the same fo sequence, so W-fragments hit WGP$/L2.
  for (int fo = 0; fo < FF / 32; ++fo) {
    // ---- GEMM1: h[16, fo*32 .. +32] = x @ W1 (full K=1024 reduction)
    v8f c0 = {}, c1 = {};
    #pragma unroll 4
    for (int kc = 0; kc < 32; ++kc) {
      v16bf a  = *(const v16bf*)&myxA[(kc * 32 + lane) * 16];
      v16bf bA = *(const v16bf*)&w1f[(((size_t)kc * 128 + fo * 2    ) * 32 + lane) * 16];
      v16bf bB = *(const v16bf*)&w1f[(((size_t)kc * 128 + fo * 2 + 1) * 32 + lane) * 16];
      c0 = __builtin_amdgcn_wmma_f32_16x16x32_bf16(false, a, false, bA, (short)0, c0, false, false);
      c1 = __builtin_amdgcn_wmma_f32_16x16x32_bf16(false, a, false, bB, (short)0, c1, false, false);
    }

    // ---- bias + ReLU, re-swizzle C-layout (n across lanes) to A-fragment
    float bi0 = bias1[fo * 32 + n];
    float bi1 = bias1[fo * 32 + 16 + n];
    #pragma unroll
    for (int r = 0; r < 8; ++r) {
      int m = mbase + r;
      float v0 = c0[r] + bi0; v0 = v0 > 0.0f ? v0 : 0.0f;
      float v1 = c1[r] + bi1; v1 = v1 > 0.0f ? v1 : 0.0f;
      hA[wave][(h0 * 16 + m) * 16 + i0] = (__bf16)v0;
      hA[wave][(h1 * 16 + m) * 16 + i1] = (__bf16)v1;
    }
    v16bf ha = *(const v16bf*)&hA[wave][lane * 16];

    // ---- GEMM2: logits[16,64] += h_tile @ W2[fo*32..+32, :]
    v16bf wb0 = *(const v16bf*)&w2f[(((size_t)fo * 4 + 0) * 32 + lane) * 16];
    v16bf wb1 = *(const v16bf*)&w2f[(((size_t)fo * 4 + 1) * 32 + lane) * 16];
    v16bf wb2 = *(const v16bf*)&w2f[(((size_t)fo * 4 + 2) * 32 + lane) * 16];
    v16bf wb3 = *(const v16bf*)&w2f[(((size_t)fo * 4 + 3) * 32 + lane) * 16];
    acc0 = __builtin_amdgcn_wmma_f32_16x16x32_bf16(false, ha, false, wb0, (short)0, acc0, false, false);
    acc1 = __builtin_amdgcn_wmma_f32_16x16x32_bf16(false, ha, false, wb1, (short)0, acc1, false, false);
    acc2 = __builtin_amdgcn_wmma_f32_16x16x32_bf16(false, ha, false, wb2, (short)0, acc2, false, false);
    acc3 = __builtin_amdgcn_wmma_f32_16x16x32_bf16(false, ha, false, wb3, (short)0, acc3, false, false);
  }

  // ---- Dump logits [16 x 64] into this wave's slab (x staging is done)
  float* lscr = (float*)myxA;
  #pragma unroll
  for (int r = 0; r < 8; ++r) {
    int m = mbase + r;
    lscr[m * EE +  0 + n] = acc0[r];
    lscr[m * EE + 16 + n] = acc1[r];
    lscr[m * EE + 32 + n] = acc2[r];
    lscr[m * EE + 48 + n] = acc3[r];
  }
  // Same-wave DS RAW -> hardware/compiler ordered; no workgroup barrier.

  // ---- Epilogue: lanes 0..15 of each wave, one token each
  if (lane < 16) {
    float l[EE];
    #pragma unroll
    for (int j = 0; j < EE; ++j)
      l[j] = lscr[lane * EE + j] + bias2[j];

    // top-1 / top-2 (first occurrence wins on ties, like jax.lax.top_k)
    int i1 = 0; float v1 = l[0];
    for (int j = 1; j < EE; ++j) if (l[j] > v1) { v1 = l[j]; i1 = j; }
    int i2 = (i1 == 0) ? 1 : 0; float v2 = l[i2];
    for (int j = 0; j < EE; ++j)
      if (j != i1 && l[j] > v2) { v2 = l[j]; i2 = j; }

    size_t t = (size_t)tokw + lane;

    // full softmax at TEMP=0.01 (subtract max for stability)
    float s = 0.0f;
    for (int j = 0; j < EE; ++j) s += __expf((l[j] - v1) * INV_TEMP);
    float inv = 1.0f / s;
    for (int j = 0; j < EE; ++j)
      __builtin_nontemporal_store(__expf((l[j] - v1) * INV_TEMP) * inv,
                                  &outOri[t * EE + j]);

    // sparse softmax over the top-2 (others are -inf -> 0)
    float e2 = __expf(v2 - v1);
    float d  = 1.0f / (1.0f + e2);
    for (int j = 0; j < EE; ++j)
      __builtin_nontemporal_store((j == i1) ? d : ((j == i2) ? e2 * d : 0.0f),
                                  &outSp[t * EE + j]);

    __builtin_nontemporal_store(i1, &outIdx[t * 2 + 0]);
    __builtin_nontemporal_store(i2, &outIdx[t * 2 + 1]);
  }
}

extern "C" void kernel_launch(void* const* d_in, const int* in_sizes, int n_in,
                              void* d_out, int out_size, void* d_ws, size_t ws_size,
                              hipStream_t stream) {
  const float* x  = (const float*)d_in[0];   // [8,4096,1024]
  const float* W1 = (const float*)d_in[1];   // [1024,2048]
  const float* b1 = (const float*)d_in[2];   // [2048]
  const float* W2 = (const float*)d_in[3];   // [2048,64]
  const float* b2 = (const float*)d_in[4];   // [64]

  __bf16* w1f = (__bf16*)d_ws;                                   // 4 MB
  __bf16* w2f = (__bf16*)((char*)d_ws + (size_t)DD * FF * 2);    // +256 KB

  float* out    = (float*)d_out;
  float* outOri = out;
  float* outSp  = out + (size_t)NTOK * EE;
  int*   outIdx = (int*)(out + (size_t)2 * NTOK * EE);

  // Pre-pack weights into bf16 WMMA B-fragment layout (read-once, lives in L2)
  pack_w_bf16<<<(DD * FF + 255) / 256, 256, 0, stream>>>(W1, w1f, DD, FF, FF / 16);
  pack_w_bf16<<<(FF * EE + 255) / 256, 256, 0, stream>>>(W2, w2f, FF, EE, EE / 16);

  // 32768 tokens / 64 per WG = 512 workgroups of 4 independent waves
  router_fused<<<NTOK / 64, 128, 0, stream>>>(x, b1, b2, w1f, w2f,
                                              outOri, outSp, outIdx);
}
